// MllamaTextCrossAttention_13297218749012
// MI455X (gfx1250) — matrix operand order repaired and checked by reading, not compiled, not run on hardware
//
#include <hip/hip_runtime.h>
#include <cstdint>

typedef __attribute__((ext_vector_type(16))) __bf16 bf16x16;
typedef __attribute__((ext_vector_type(2)))  __bf16 bf16x2;
typedef __attribute__((ext_vector_type(8)))  float  f32x8;

struct alignas(16) U4 { unsigned x, y, z, w; };
struct alignas(16) F4 { float x, y, z, w; };

union AFrag { bf16x16 v; unsigned u[8]; };

__device__ __forceinline__ unsigned short f2bf(float f) {
    unsigned u = __float_as_uint(f);
    u += 0x7FFFu + ((u >> 16) & 1u);   // round-to-nearest-even
    return (unsigned short)(u >> 16);
}
__device__ __forceinline__ unsigned packbf(float a, float b) {
#if __has_builtin(__builtin_amdgcn_cvt_pk_bf16_f32)
    union { bf16x2 v; unsigned u; } x;
    x.v = __builtin_amdgcn_cvt_pk_bf16_f32(a, b);
    return x.u;
#else
    return (unsigned)f2bf(a) | ((unsigned)f2bf(b) << 16);
#endif
}

// ---------------------------------------------------------------------------
// NT GEMM: C[M,N] = A[M,K] * B[N,K]^T  (A fp32 or bf16, B fp32 weights)
// Block tile 128x128, 8 waves (4 in M x 2 in N), each wave 32x64 = 2x4 tiles.
// Double-buffered LDS + register prefetch. Loads use clamped row indices
// (no divergent guards); only the epilogue stores are bounds-checked.
// ---------------------------------------------------------------------------
template <bool ABF16>
__global__ __launch_bounds__(256, 1) void gemm_nt(const void* __restrict__ Ap,
                                                  const float* __restrict__ Bw,
                                                  float* __restrict__ C,
                                                  int M, int N, int K) {
    __shared__ alignas(16) unsigned ldsA[2][128 * 16];  // 2 x 8 KB, A-frag layout
    __shared__ alignas(16) unsigned ldsB[2][128 * 16];  // 2 x 8 KB, B-frag layout
    const int tid  = threadIdx.x;
    const int lane = tid & 31, wid = tid >> 5;
    const int lh = lane >> 4, lr = lane & 15;
    const int bm = blockIdx.y * 128, bn = blockIdx.x * 128;
    const int wm = wid & 3, wn = wid >> 2;

    F4 ra[2][2], rb[2][2];
    U4 rha[2];

    auto gload = [&](int kk) {
#pragma unroll
        for (int c = 0; c < 2; ++c) {
            int id = tid + c * 256;
            int m = id >> 2, q = id & 3;
            int gm = bm + m;  gm = (gm < M) ? gm : (M - 1);   // clamp, no branch
            int gn = bn + m;  gn = (gn < N) ? gn : (N - 1);
            if constexpr (ABF16) {
                rha[c] = *(const U4*)((const unsigned short*)Ap + (size_t)gm * K + kk + q * 8);
            } else {
                const float* src = (const float*)Ap + (size_t)gm * K + kk + q * 8;
                ra[c][0] = *(const F4*)src;
                ra[c][1] = *(const F4*)(src + 4);
            }
            const float* bsrc = Bw + (size_t)gn * K + kk + q * 8;
            rb[c][0] = *(const F4*)bsrc;
            rb[c][1] = *(const F4*)(bsrc + 4);
        }
    };
    auto lstore = [&](int buf) {
#pragma unroll
        for (int c = 0; c < 2; ++c) {
            int id = tid + c * 256;
            int m = id >> 2, q = id & 3;
            U4 av;
            if constexpr (ABF16) {
                av = rha[c];
            } else {
                av = U4{packbf(ra[c][0].x, ra[c][0].y), packbf(ra[c][0].z, ra[c][0].w),
                        packbf(ra[c][1].x, ra[c][1].y), packbf(ra[c][1].z, ra[c][1].w)};
            }
            *(U4*)&ldsA[buf][m * 16 + (q & 1) * 8 + (q >> 1) * 4] = av;
            U4 bv = {packbf(rb[c][0].x, rb[c][0].y), packbf(rb[c][0].z, rb[c][0].w),
                     packbf(rb[c][1].x, rb[c][1].y), packbf(rb[c][1].z, rb[c][1].w)};
            *(U4*)&ldsB[buf][m * 16 + (q & 1) * 8 + (q >> 1) * 4] = bv;
        }
    };

    f32x8 acc[2][4] = {};
    const int nk = K >> 5;
    gload(0);
    lstore(0);
    __syncthreads();

#pragma unroll 1
    for (int it = 0; it < nk; ++it) {
        const int buf = it & 1;
        if (it + 1 < nk) gload((it + 1) << 5);  // prefetch overlaps the WMMAs

        AFrag af[2], bf[4];
#pragma unroll
        for (int i = 0; i < 2; ++i) {
            int m = wm * 32 + i * 16 + lr;
            *(U4*)&af[i].u[0] = *(const U4*)&ldsA[buf][m * 16 + lh * 8];
            *(U4*)&af[i].u[4] = *(const U4*)&ldsA[buf][m * 16 + lh * 8 + 4];
        }
#pragma unroll
        for (int j = 0; j < 4; ++j) {
            int n = wn * 64 + j * 16 + lr;
            *(U4*)&bf[j].u[0] = *(const U4*)&ldsB[buf][n * 16 + lh * 8];
            *(U4*)&bf[j].u[4] = *(const U4*)&ldsB[buf][n * 16 + lh * 8 + 4];
        }
#pragma unroll
        for (int i = 0; i < 2; ++i)
#pragma unroll
            for (int j = 0; j < 4; ++j)
                acc[i][j] = __builtin_amdgcn_wmma_f32_16x16x32_bf16(
                    false, af[i].v, false, bf[j].v, (short)0, acc[i][j], false, false);

        if (it + 1 < nk) lstore(buf ^ 1);
        __syncthreads();
    }

    // epilogue: C layout = lane column n, VGPR r -> row r + 8*laneHalf
#pragma unroll
    for (int i = 0; i < 2; ++i)
        for (int j = 0; j < 4; ++j) {
            int n = bn + wn * 64 + j * 16 + lr;
            if (n < N)
                for (int r = 0; r < 8; ++r) {
                    int m = bm + wm * 32 + i * 16 + r + 8 * lh;
                    if (m < M) C[(size_t)m * N + n] = acc[i][j][r];
                }
        }
}

// ---------------------------------------------------------------------------
// Per-(token,head) RMSNorm over head_dim=128 + fp32->bf16 convert.
// One wave per row. do_norm==0 => plain convert (for V).
// ---------------------------------------------------------------------------
__global__ __launch_bounds__(256) void rms_cvt(const float* __restrict__ x,
                                               const float* __restrict__ w,
                                               unsigned short* __restrict__ y,
                                               int rows, int do_norm) {
    int row = blockIdx.x * 8 + (threadIdx.x >> 5);
    if (row >= rows) return;
    int lane = threadIdx.x & 31;
    F4 v = *(const F4*)(x + (size_t)row * 128 + lane * 4);
    float r = 1.0f;
    F4 wv = {1.f, 1.f, 1.f, 1.f};
    if (do_norm) {
        float s = v.x * v.x + v.y * v.y + v.z * v.z + v.w * v.w;
#pragma unroll
        for (int m = 16; m >= 1; m >>= 1) s += __shfl_xor(s, m, 32);
        r = rsqrtf(s * (1.0f / 128.0f) + 1e-5f);
        wv = *(const F4*)(w + lane * 4);
    }
    unsigned* dst = (unsigned*)(y + (size_t)row * 128 + lane * 4);
    dst[0] = packbf(v.x * r * wv.x, v.y * r * wv.y);
    dst[1] = packbf(v.z * r * wv.z, v.w * r * wv.w);
}

// ---------------------------------------------------------------------------
// GQA flash attention. Grid: (qtile=4, head=32, batch=4). Block: 128 (4 waves),
// each wave owns 16 q rows of the 64-row WG tile. KV chunks of 32.
// K/V double-buffered in LDS with register prefetch whose live range ends at
// the mid-iteration barrier (prefetch regs dead before the PV section).
// Q fragments are re-read from LDS (never overwritten) to cap VGPR pressure.
// ---------------------------------------------------------------------------
__global__ __launch_bounds__(128, 1) void attn_fwd(const unsigned short* __restrict__ qb,
                                                   const unsigned short* __restrict__ kb,
                                                   const unsigned short* __restrict__ vb,
                                                   unsigned short* __restrict__ ob) {
    __shared__ alignas(16) unsigned ldsQ[64 * 64];        // 16 KB  A-frag layout
    __shared__ alignas(16) unsigned ldsK[2][32 * 64];     // 2x8 KB B-frag layout (k=d)
    __shared__ alignas(16) unsigned ldsV[2][128 * 16];    // 2x8 KB B-frag layout (k=kc)
    __shared__ alignas(16) unsigned short ldsP[4][16 * 32];  // 4 KB, per-wave

    const int tid = threadIdx.x;
    const int lane = tid & 31, wid = tid >> 5;
    const int lh = lane >> 4, lr = lane & 15;
    const int b = blockIdx.z, h = blockIdx.y, qt = blockIdx.x;
    const int kvh = h >> 2;
    const int q0 = b * 256 + qt * 64;

    // stage Q tile (64 rows x 128 d), already bf16
#pragma unroll
    for (int c = 0; c < 8; ++c) {
        int id = tid + c * 128;
        int m = id >> 4, q = id & 15;
        int ks = q >> 2, rr = q & 3;
        const U4* src = (const U4*)(qb + (size_t)(q0 + m) * 4096 + h * 128 + q * 8);
        *(U4*)&ldsQ[m * 64 + ks * 16 + (rr & 1) * 8 + (rr >> 1) * 4] = *src;
    }

    const int qrow = (wid * 16 + lr) * 64 + lh * 8;  // base for this lane's Q frags

    // K/V register prefetch buffers (4 chunks each of 8 bf16)
    U4 rk[4], rv[4];
    auto gloadKV = [&](int kc0) {
#pragma unroll
        for (int c = 0; c < 4; ++c) {
            int id = tid + c * 128;
            int n = id >> 4, q = id & 15;
            int kc = kc0 + n;
            kc = (kc < 1601) ? kc : 1600;  // clamp; masked cols die in softmax
            size_t row = (size_t)(b * 1601 + kc) * 1024 + kvh * 128 + q * 8;
            rk[c] = *(const U4*)(kb + row);
            rv[c] = *(const U4*)(vb + row);
        }
    };
    auto lstoreKV = [&](int buf) {
#pragma unroll
        for (int c = 0; c < 4; ++c) {
            int id = tid + c * 128;
            int n = id >> 4, q = id & 15;
            int ks = q >> 2, rr = q & 3;
            *(U4*)&ldsK[buf][n * 64 + ks * 16 + (rr & 1) * 8 + (rr >> 1) * 4] = rk[c];
            const unsigned short* ve = (const unsigned short*)&rv[c];
            unsigned short* vd = (unsigned short*)ldsV[buf];
#pragma unroll
            for (int e = 0; e < 8; ++e) {
                int d = q * 8 + e;
                vd[d * 32 + ((n >> 3) & 1) * 16 + (n >> 4) * 8 + (n & 7)] = ve[e];
            }
        }
    };

    float rowmax[8], rowsum[8];
    f32x8 o[8] = {};
#pragma unroll
    for (int r = 0; r < 8; ++r) { rowmax[r] = -1e30f; rowsum[r] = 0.f; }

    const float scale = 0.08838834764831845f;  // 128^-0.5
    const int nchunk = 51;                     // ceil(1601/32)

    gloadKV(0);
    lstoreKV(0);
    __syncthreads();

#pragma unroll 1
    for (int it = 0; it < nchunk; ++it) {
        const int buf = it & 1;
        const int kc0 = it * 32;
        if (it + 1 < nchunk) gloadKV(kc0 + 32);  // overlaps S-WMMAs + softmax

        // S = Q * K^T  (2 kc tiles, 4 k-steps over d); Q frag re-read from LDS
        f32x8 sacc[2] = {};
#pragma unroll
        for (int ks = 0; ks < 4; ++ks) {
            AFrag qa, kf0, kf1;
            *(U4*)&qa.u[0] = *(const U4*)&ldsQ[qrow + ks * 16];
            *(U4*)&qa.u[4] = *(const U4*)&ldsQ[qrow + ks * 16 + 4];
            *(U4*)&kf0.u[0] = *(const U4*)&ldsK[buf][lr * 64 + ks * 16 + lh * 8];
            *(U4*)&kf0.u[4] = *(const U4*)&ldsK[buf][lr * 64 + ks * 16 + lh * 8 + 4];
            *(U4*)&kf1.u[0] = *(const U4*)&ldsK[buf][(16 + lr) * 64 + ks * 16 + lh * 8];
            *(U4*)&kf1.u[4] = *(const U4*)&ldsK[buf][(16 + lr) * 64 + ks * 16 + lh * 8 + 4];
            sacc[0] = __builtin_amdgcn_wmma_f32_16x16x32_bf16(
                false, qa.v, false, kf0.v, (short)0, sacc[0], false, false);
            sacc[1] = __builtin_amdgcn_wmma_f32_16x16x32_bf16(
                false, qa.v, false, kf1.v, (short)0, sacc[1], false, false);
        }

        // mask + online softmax (column index is per-lane uniform)
        float cmax[8], corr[8], psum[8];
        int c0 = kc0 + lr, c1 = kc0 + 16 + lr;
#pragma unroll
        for (int r = 0; r < 8; ++r) {
            float a0 = (c0 < 1601) ? sacc[0][r] * scale : -1e30f;
            float a1 = (c1 < 1601) ? sacc[1][r] * scale : -1e30f;
            sacc[0][r] = a0; sacc[1][r] = a1;
            cmax[r] = fmaxf(a0, a1);
        }
#pragma unroll
        for (int msk = 8; msk >= 1; msk >>= 1)
#pragma unroll
            for (int r = 0; r < 8; ++r)
                cmax[r] = fmaxf(cmax[r], __shfl_xor(cmax[r], msk, 32));
#pragma unroll
        for (int r = 0; r < 8; ++r) {
            float mn = fmaxf(rowmax[r], cmax[r]);
            corr[r] = __expf(rowmax[r] - mn);
            rowmax[r] = mn;
            float p0 = __expf(sacc[0][r] - mn);
            float p1 = __expf(sacc[1][r] - mn);
            sacc[0][r] = p0; sacc[1][r] = p1;
            psum[r] = p0 + p1;
        }
#pragma unroll
        for (int msk = 8; msk >= 1; msk >>= 1)
#pragma unroll
            for (int r = 0; r < 8; ++r)
                psum[r] += __shfl_xor(psum[r], msk, 32);
#pragma unroll
        for (int r = 0; r < 8; ++r) rowsum[r] = rowsum[r] * corr[r] + psum[r];

        // P: C-layout -> A-frag layout via per-wave LDS region
#pragma unroll
        for (int tn = 0; tn < 2; ++tn)
#pragma unroll
            for (int r = 0; r < 8; ++r) {
                int m = r + 8 * lh;
                int kc = tn * 16 + lr;
                int hh, cc, ss;
                if (kc < 16) { hh = kc >> 3; cc = (kc & 7) >> 1; ss = kc & 1; }
                else { int k2 = kc - 16; hh = k2 >> 3; cc = 4 + ((k2 & 7) >> 1); ss = k2 & 1; }
                ldsP[wid][(m * 16 + hh * 8 + cc) * 2 + ss] = f2bf(sacc[tn][r]);
            }

        // drain the prefetch into the other buffer NOW: prefetch registers are
        // dead before the register-hungry PV section begins.
        if (it + 1 < nchunk) lstoreKV(buf ^ 1);
        __syncthreads();

        AFrag pf;
        const unsigned* pw = (const unsigned*)ldsP[wid];
        *(U4*)&pf.u[0] = *(const U4*)&pw[lr * 16 + lh * 8];
        *(U4*)&pf.u[4] = *(const U4*)&pw[lr * 16 + lh * 8 + 4];

        // O += P * V  (8 d tiles, paired fragment loads)
#pragma unroll
        for (int jj = 0; jj < 4; ++jj) {
            int j0 = jj * 2, j1 = jj * 2 + 1;
            AFrag vf0, vf1;
            *(U4*)&vf0.u[0] = *(const U4*)&ldsV[buf][(j0 * 16 + lr) * 16 + lh * 8];
            *(U4*)&vf0.u[4] = *(const U4*)&ldsV[buf][(j0 * 16 + lr) * 16 + lh * 8 + 4];
            *(U4*)&vf1.u[0] = *(const U4*)&ldsV[buf][(j1 * 16 + lr) * 16 + lh * 8];
            *(U4*)&vf1.u[4] = *(const U4*)&ldsV[buf][(j1 * 16 + lr) * 16 + lh * 8 + 4];
#pragma unroll
            for (int r = 0; r < 8; ++r) { o[j0][r] *= corr[r]; o[j1][r] *= corr[r]; }
            o[j0] = __builtin_amdgcn_wmma_f32_16x16x32_bf16(
                false, pf.v, false, vf0.v, (short)0, o[j0], false, false);
            o[j1] = __builtin_amdgcn_wmma_f32_16x16x32_bf16(
                false, pf.v, false, vf1.v, (short)0, o[j1], false, false);
        }
        __syncthreads();
    }

    float inv[8];
#pragma unroll
    for (int r = 0; r < 8; ++r) inv[r] = 1.0f / rowsum[r];
#pragma unroll
    for (int j = 0; j < 8; ++j)
#pragma unroll
        for (int r = 0; r < 8; ++r) {
            int token = q0 + wid * 16 + r + 8 * lh;
            int d = j * 16 + lr;
            ob[(size_t)token * 4096 + h * 128 + d] = f2bf(o[j][r] * inv[r]);
        }
}

// ---------------------------------------------------------------------------
extern "C" void kernel_launch(void* const* d_in, const int* in_sizes, int n_in,
                              void* d_out, int out_size, void* d_ws, size_t ws_size,
                              hipStream_t stream) {
    (void)in_sizes; (void)n_in; (void)out_size; (void)ws_size;
    const float* hidden = (const float*)d_in[0];
    const float* cross  = (const float*)d_in[1];
    const float* Wq = (const float*)d_in[2];
    const float* Wk = (const float*)d_in[3];
    const float* Wv = (const float*)d_in[4];
    const float* Wo = (const float*)d_in[5];
    const float* qnw = (const float*)d_in[6];
    const float* knw = (const float*)d_in[7];
    float* out = (float*)d_out;

    const int Tq = 1024, Tk = 6404, HID = 4096, KVD = 1024;
    char* ws = (char*)d_ws;
    size_t off = 0;
    auto carve = [&](size_t bytes) {
        char* p = ws + off;
        off = (off + bytes + 255) & ~(size_t)255;
        return p;
    };
    float* qf = (float*)carve((size_t)Tq * HID * 4);
    float* kf = (float*)carve((size_t)Tk * KVD * 4);
    float* vf = (float*)carve((size_t)Tk * KVD * 4);
    unsigned short* qb = (unsigned short*)carve((size_t)Tq * HID * 2);
    unsigned short* kb = (unsigned short*)carve((size_t)Tk * KVD * 2);
    unsigned short* vb = (unsigned short*)carve((size_t)Tk * KVD * 2);
    unsigned short* ab = (unsigned short*)carve((size_t)Tq * HID * 2);

    dim3 blk(256);
    gemm_nt<false><<<dim3(32, 8),  blk, 0, stream>>>(hidden, Wq, qf, Tq, HID, HID);
    gemm_nt<false><<<dim3(8, 51),  blk, 0, stream>>>(cross,  Wk, kf, Tk, KVD, HID);
    gemm_nt<false><<<dim3(8, 51),  blk, 0, stream>>>(cross,  Wv, vf, Tk, KVD, HID);
    rms_cvt<<<dim3((Tq * 32 + 7) / 8), blk, 0, stream>>>(qf, qnw, qb, Tq * 32, 1);
    rms_cvt<<<dim3((Tk * 8 + 7) / 8),  blk, 0, stream>>>(kf, knw, kb, Tk * 8, 1);
    rms_cvt<<<dim3((Tk * 8 + 7) / 8),  blk, 0, stream>>>(vf, nullptr, vb, Tk * 8, 0);
    attn_fwd<<<dim3(4, 32, 4), dim3(128), 0, stream>>>(qb, kb, vb, ab);
    gemm_nt<true><<<dim3(32, 8), blk, 0, stream>>>(ab, Wo, out, Tq, HID, HID);
}
